// GCN_77661598646285
// MI455X (gfx1250) — compile-verified, matrix operand order checked by reading, not run on hardware
//
#include <hip/hip_runtime.h>
#include <hip/hip_bf16.h>
#include <math.h>

typedef __attribute__((ext_vector_type(16))) _Float16 v16h;
typedef __attribute__((ext_vector_type(8)))  float    v8f;

#define FEAT 128          // K dimension for every GEMM
#define NCLS 40
#define NCLS_PAD 48

// ---------------- W prep: fp32 [K][N] row-major -> f16 [Npad][K] (transposed) ----
__global__ void prep_w_kernel(const float* __restrict__ W, _Float16* __restrict__ Wt,
                              int N, int Npad) {
    int idx = blockIdx.x * blockDim.x + threadIdx.x;
    if (idx >= Npad * FEAT) return;
    int n = idx / FEAT;
    int k = idx - n * FEAT;
    float v = (n < N) ? W[(size_t)k * N + n] : 0.0f;
    Wt[(size_t)n * FEAT + k] = (_Float16)v;
}

// ---------------- degree / dinv ------------------------------------------------
__global__ void init_deg_kernel(float* deg, int M) {
    int i = blockIdx.x * blockDim.x + threadIdx.x;
    if (i < M) deg[i] = 1.0f;                    // self-loop
}
__global__ void count_deg_kernel(const int* __restrict__ dst, float* deg, int E) {
    int e = blockIdx.x * blockDim.x + threadIdx.x;
    if (e < E) atomicAdd(&deg[dst[e]], 1.0f);
}
__global__ void dinv_kernel(const float* __restrict__ deg, float* dinv, int M) {
    int i = blockIdx.x * blockDim.x + threadIdx.x;
    if (i < M) dinv[i] = rsqrtf(deg[i]);         // deg >= 1 always
}

// ---------------- WMMA GEMM: C[M,NOut] = (relu?)A[M,128] * W[128,NOut] ---------
// Wt is f16 [Npad][128] (N-major, K contiguous). NT = Npad/16 tiles per wave.
// One wave computes a 16-row block across all N tiles; K loop of 4 x 32.
template <int NT>
__global__ void gemm_wmma_kernel(const float* __restrict__ A,
                                 const _Float16* __restrict__ Wt,
                                 float* __restrict__ C,
                                 int M, int NOut, int applyRelu) {
    const int lane = threadIdx.x & 31;
    const int wave = threadIdx.x >> 5;
    int m0 = (blockIdx.x * (blockDim.x >> 5) + wave) << 4;

    // clamp row for loads so EXEC stays all-ones into WMMA; stores are guarded
    int mrow = m0 + (lane & 15);
    if (mrow >= M) mrow = M - 1;
    const int khalf = (lane >> 4) << 4;          // 0 or 16 within a K=32 step

    v8f zero = {};
    v8f acc[NT];
#pragma unroll
    for (int i = 0; i < NT; ++i) acc[i] = zero;

#pragma unroll
    for (int kt = 0; kt < 4; ++kt) {
        const int kbase = kt * 32 + khalf;
        const float* ap = A + (size_t)mrow * FEAT + kbase;
        v16h a;
#pragma unroll
        for (int i = 0; i < 16; ++i) {
            float v = ap[i];
            if (applyRelu) v = fmaxf(v, 0.0f);
            a[i] = (_Float16)v;
        }
#pragma unroll
        for (int nt = 0; nt < NT; ++nt) {
            const v16h b = *(const v16h*)(Wt + (size_t)(nt * 16 + (lane & 15)) * FEAT + kbase);
            acc[nt] = __builtin_amdgcn_wmma_f32_16x16x32_f16(
                false, a, false, b, (short)0, acc[nt], false, false);
        }
    }

    // C/D layout: VGPR r -> M = m0 + r + 8*lane[4], N = nt*16 + lane[3:0]
    const int mr = m0 + ((lane >> 4) << 3);
#pragma unroll
    for (int nt = 0; nt < NT; ++nt) {
        int n = nt * 16 + (lane & 15);
        if (n < NOut) {
#pragma unroll
            for (int r = 0; r < 8; ++r) {
                int m = mr + r;
                if (m < M) C[(size_t)m * NOut + n] = acc[nt][r];
            }
        }
    }
}

// ---------------- out[i][f] = b[f] --------------------------------------------
__global__ void init_bias_kernel(float* __restrict__ out, const float* __restrict__ b,
                                 long total, int F) {
    long i = (long)blockIdx.x * blockDim.x + threadIdx.x;
    if (i < total) out[i] = b[i % F];
}

// ---------------- edge + self-loop scatter: out[d] += coef * h[s] -------------
// one wave per (edge or self-loop); lane handles 4 features.
__global__ void aggregate_kernel(const float* __restrict__ h, float* __restrict__ out,
                                 const int* __restrict__ src, const int* __restrict__ dst,
                                 const float* __restrict__ dinv,
                                 int E, int M, int F) {
    const int lane = threadIdx.x & 31;
    const int wave = threadIdx.x >> 5;
    long e = (long)blockIdx.x * (blockDim.x >> 5) + wave;
    long ET = (long)E + M;
    if (e >= ET) return;
    int s, d;
    if (e < (long)E) { s = src[e]; d = dst[e]; }
    else             { s = d = (int)(e - E); }   // self loop
    float c = dinv[s] * dinv[d];
    int f = lane * 4;
    if (f < F) {
        const float4 hv = *(const float4*)(h + (size_t)s * F + f);
        float* op = out + (size_t)d * F + f;
        atomicAdd(op + 0, hv.x * c);
        atomicAdd(op + 1, hv.y * c);
        atomicAdd(op + 2, hv.z * c);
        atomicAdd(op + 3, hv.w * c);
    }
}

// ---------------- row-wise log_softmax over F=40 (one wave per row) -----------
__global__ void log_softmax_kernel(float* __restrict__ out, int M, int F) {
    const int lane = threadIdx.x & 31;
    const int wave = threadIdx.x >> 5;
    int row = blockIdx.x * (blockDim.x >> 5) + wave;
    if (row >= M) return;
    float* p = out + (size_t)row * F;
    float v0 = (lane < F)      ? p[lane]      : -INFINITY;
    float v1 = (lane + 32 < F) ? p[lane + 32] : -INFINITY;
    float mx = fmaxf(v0, v1);
    for (int off = 16; off > 0; off >>= 1) mx = fmaxf(mx, __shfl_xor(mx, off, 32));
    float s = 0.0f;
    if (lane < F)      s += expf(v0 - mx);
    if (lane + 32 < F) s += expf(v1 - mx);
    for (int off = 16; off > 0; off >>= 1) s += __shfl_xor(s, off, 32);
    float lse = mx + logf(s);
    if (lane < F)      p[lane]      = v0 - lse;
    if (lane + 32 < F) p[lane + 32] = v1 - lse;
}

extern "C" void kernel_launch(void* const* d_in, const int* in_sizes, int n_in,
                              void* d_out, int out_size, void* d_ws, size_t ws_size,
                              hipStream_t stream) {
    const float* x  = (const float*)d_in[0];
    const int*   ei = (const int*)  d_in[1];
    const float* W1 = (const float*)d_in[2];
    const float* b1 = (const float*)d_in[3];
    const float* W2 = (const float*)d_in[4];
    const float* b2 = (const float*)d_in[5];
    const float* W3 = (const float*)d_in[6];
    const float* b3 = (const float*)d_in[7];

    const int M = in_sizes[0] / FEAT;       // 100000 nodes
    const int E = in_sizes[1] / 2;          // 1600000 edges
    const int* src = ei;
    const int* dst = ei + E;

    // -------- workspace partition (256-byte aligned) --------
    char* ws = (char*)d_ws;
    size_t off = 0;
    auto alloc = [&](size_t bytes) -> void* {
        void* p = ws + off;
        off += (bytes + 255) & ~(size_t)255;
        return p;
    };
    float*     bufA = (float*)    alloc((size_t)M * FEAT * sizeof(float));
    float*     bufB = (float*)    alloc((size_t)M * FEAT * sizeof(float));
    float*     deg  = (float*)    alloc((size_t)M * sizeof(float));
    float*     dinv = (float*)    alloc((size_t)M * sizeof(float));
    _Float16*  Wt1  = (_Float16*) alloc((size_t)FEAT * FEAT * sizeof(_Float16));
    _Float16*  Wt2  = (_Float16*) alloc((size_t)FEAT * FEAT * sizeof(_Float16));
    _Float16*  Wt3  = (_Float16*) alloc((size_t)NCLS_PAD * FEAT * sizeof(_Float16));
    float*     outF = (float*)d_out;

    // -------- weight prep --------
    {
        int t = FEAT * FEAT;
        prep_w_kernel<<<(t + 255) / 256, 256, 0, stream>>>(W1, Wt1, FEAT, FEAT);
        prep_w_kernel<<<(t + 255) / 256, 256, 0, stream>>>(W2, Wt2, FEAT, FEAT);
        int t3 = NCLS_PAD * FEAT;
        prep_w_kernel<<<(t3 + 255) / 256, 256, 0, stream>>>(W3, Wt3, NCLS, NCLS_PAD);
    }

    // -------- degrees & normalization --------
    init_deg_kernel<<<(M + 255) / 256, 256, 0, stream>>>(deg, M);
    count_deg_kernel<<<(E + 255) / 256, 256, 0, stream>>>(dst, deg, E);
    dinv_kernel<<<(M + 255) / 256, 256, 0, stream>>>(deg, dinv, M);

    const int gemmBlocks = (M + 127) / 128;         // 8 waves x 16 rows per block
    const long ET = (long)E + M;
    const int aggBlocks = (int)((ET + 7) / 8);      // 8 waves per block, 1 edge/wave
    const long MF = (long)M * FEAT;
    const long MC = (long)M * NCLS;

    // -------- layer 1: h1 = x W1 ; out = scatter(coef*h1) + b1 (relu deferred) --
    gemm_wmma_kernel<8><<<gemmBlocks, 256, 0, stream>>>(x, Wt1, bufA, M, FEAT, 0);
    init_bias_kernel<<<(int)((MF + 255) / 256), 256, 0, stream>>>(bufB, b1, MF, FEAT);
    aggregate_kernel<<<aggBlocks, 256, 0, stream>>>(bufA, bufB, src, dst, dinv, E, M, FEAT);

    // -------- layer 2 (relu fused into A-load conversion) --------
    gemm_wmma_kernel<8><<<gemmBlocks, 256, 0, stream>>>(bufB, Wt2, bufA, M, FEAT, 1);
    init_bias_kernel<<<(int)((MF + 255) / 256), 256, 0, stream>>>(bufB, b2, MF, FEAT);
    aggregate_kernel<<<aggBlocks, 256, 0, stream>>>(bufA, bufB, src, dst, dinv, E, M, FEAT);

    // -------- layer 3: N=40 (padded to 48 -> 3 WMMA tiles) --------
    gemm_wmma_kernel<3><<<gemmBlocks, 256, 0, stream>>>(bufB, Wt3, bufA, M, NCLS, 1);
    init_bias_kernel<<<(int)((MC + 255) / 256), 256, 0, stream>>>(outF, b3, MC, NCLS);
    aggregate_kernel<<<aggBlocks, 256, 0, stream>>>(bufA, outF, src, dst, dinv, E, M, NCLS);

    // -------- log softmax --------
    log_softmax_kernel<<<(M + 7) / 8, 256, 0, stream>>>(outF, M, NCLS);
}
